// MeshConv_37641093382405
// MI455X (gfx1250) — matrix operand order repaired and checked by reading, not compiled, not run on hardware
//
#include <hip/hip_runtime.h>

typedef __attribute__((ext_vector_type(16))) __bf16 v16bf;
typedef __attribute__((ext_vector_type(8)))  __bf16 v8bf;
typedef __attribute__((ext_vector_type(8)))  float  v8f;

union ABFrag { v16bf v; v8bf h[2]; };

#define OUT_CH   128
#define K_TOT    320          // 5 taps * 64 in_ch, k = kk*64 + c
#define A_STRIDE 328          // padded row stride (bf16 elems), 16B aligned
#define M_TILE   64

// ws layout:
//   [0, 81920)        bf16 Wb[128][320]   (W transposed to [o][k])
//   [81920, +512)     float gsum[128]
//   [82432, +512)     float gsq [128]
//   [82944, +512)     float scale[128]
//   [83456, +512)     float shift[128]

__global__ __launch_bounds__(256) void prep_kernel(const float* __restrict__ W,
                                                   __bf16* __restrict__ Wb,
                                                   float* __restrict__ stats) {
  int i = blockIdx.x * 256 + threadIdx.x;
  if (blockIdx.x == 0) stats[threadIdx.x] = 0.0f;   // zero gsum[128] + gsq[128]
  if (i < OUT_CH * K_TOT) {
    int o = i / K_TOT;
    int k = i - o * K_TOT;
    int kk = k >> 6, c = k & 63;
    // W is [o][c][kk] row-major: idx = o*320 + c*5 + kk
    Wb[i] = (__bf16)W[o * K_TOT + c * 5 + kk];
  }
}

__global__ __launch_bounds__(256) void conv_kernel(const float* __restrict__ x,
                                                   const int*   __restrict__ e2e,
                                                   const __bf16* __restrict__ Wb,
                                                   float* __restrict__ out,
                                                   float* __restrict__ gsum,
                                                   float* __restrict__ gsq) {
  __shared__ int rowidx[M_TILE * 5];
  __shared__ __align__(16) __bf16 As[M_TILE * A_STRIDE];
  __shared__ float lsum[OUT_CH];
  __shared__ float lsq[OUT_CH];

  const int tid = threadIdx.x;
  const int e0  = blockIdx.x * M_TILE;

  if (tid < OUT_CH) { lsum[tid] = 0.0f; lsq[tid] = 0.0f; }

  // Row indices for the gather: self + 4 neighbors per edge
  for (int i = tid; i < M_TILE * 5; i += 256) {
    int m = i / 5, kk = i - m * 5;
    rowidx[i] = (kk == 0) ? (e0 + m) : e2e[(long)(e0 + m) * 4 + (kk - 1)];
  }
  __syncthreads();

  // Gather A tile [64 x 320] f32 -> bf16 into LDS (coalesced 64-float runs)
  for (int i = tid; i < M_TILE * K_TOT; i += 256) {
    int m = i / K_TOT, k = i - m * K_TOT;
    int kk = k >> 6, c = k & 63;
    int row = rowidx[m * 5 + kk];
    As[m * A_STRIDE + k] = (__bf16)x[(long)row * 64 + c];
  }
  __syncthreads();

  const int lane = tid & 31;
  const int wave = tid >> 5;
  const int nn   = lane & 15;   // column within 16-wide tile
  const int h    = lane >> 4;   // half-wave
  const int n0   = wave * 16;   // this wave's N strip

  v8f zero = {};
  v8f acc[4];
  #pragma unroll
  for (int ms = 0; ms < 4; ++ms) acc[ms] = zero;

  const __bf16* wrow = Wb + (long)(n0 + nn) * K_TOT;

  #pragma unroll 2
  for (int kb = 0; kb < K_TOT; kb += 32) {
    // B frag (32x16): lane holds column n; K runs [kb+h*8, +8) and [kb+16+h*8, +8)
    ABFrag b;
    b.h[0] = *(const v8bf*)(wrow + kb + h * 8);
    b.h[1] = *(const v8bf*)(wrow + kb + 16 + h * 8);
    #pragma unroll
    for (int ms = 0; ms < 4; ++ms) {
      // A frag (16x32): lane holds row m; K runs [kb+h*8, +8) and [kb+16+h*8, +8)
      ABFrag a;
      const __bf16* arow = &As[(ms * 16 + nn) * A_STRIDE + kb];
      a.h[0] = *(const v8bf*)(arow + h * 8);
      a.h[1] = *(const v8bf*)(arow + 16 + h * 8);
      acc[ms] = __builtin_amdgcn_wmma_f32_16x16x32_bf16(
          false, a.v, false, b.v, (short)0, acc[ms], false, false);
    }
  }

  // C layout: VGPR r -> M = ms*16 + r + h*8, N = n0 + nn (fixed per lane)
  float s = 0.0f, s2 = 0.0f;
  #pragma unroll
  for (int ms = 0; ms < 4; ++ms) {
    int mbase = e0 + ms * 16 + h * 8;
    #pragma unroll
    for (int r = 0; r < 8; ++r) {
      float v = acc[ms][r];
      out[(long)(mbase + r) * OUT_CH + (n0 + nn)] = v;
      s  += v;
      s2 += v * v;
    }
  }
  atomicAdd(&lsum[n0 + nn], s);
  atomicAdd(&lsq [n0 + nn], s2);
  __syncthreads();
  if (tid < OUT_CH) atomicAdd(&gsum[tid], lsum[tid]);
  else              atomicAdd(&gsq[tid - OUT_CH], lsq[tid - OUT_CH]);
}

__global__ void stats_kernel(const float* __restrict__ gsum,
                             const float* __restrict__ gsq,
                             const float* __restrict__ gamma,
                             const float* __restrict__ beta,
                             float* __restrict__ scale,
                             float* __restrict__ shift,
                             float invE) {
  int o = threadIdx.x;
  float mu  = gsum[o] * invE;
  float var = gsq[o] * invE - mu * mu;       // biased variance, matches jnp.var
  float sc  = gamma[o] * rsqrtf(var + 1e-5f);
  scale[o] = sc;
  shift[o] = beta[o] - mu * sc;
}

__global__ __launch_bounds__(256) void bnrelu_kernel(float4* __restrict__ out4,
                                                     const float* __restrict__ scale,
                                                     const float* __restrict__ shift) {
  long i = (long)blockIdx.x * 256 + threadIdx.x;   // over E*128/4 float4s
  int c0 = ((int)(i & 31)) * 4;                    // 128 ch / 4 = 32 float4 per row
  float4 v = out4[i];
  v.x = fmaxf(v.x * scale[c0 + 0] + shift[c0 + 0], 0.0f);
  v.y = fmaxf(v.y * scale[c0 + 1] + shift[c0 + 1], 0.0f);
  v.z = fmaxf(v.z * scale[c0 + 2] + shift[c0 + 2], 0.0f);
  v.w = fmaxf(v.w * scale[c0 + 3] + shift[c0 + 3], 0.0f);
  out4[i] = v;
}

extern "C" void kernel_launch(void* const* d_in, const int* in_sizes, int n_in,
                              void* d_out, int out_size, void* d_ws, size_t ws_size,
                              hipStream_t stream) {
  const float* x     = (const float*)d_in[0];
  const int*   e2e   = (const int*)  d_in[1];
  const float* W     = (const float*)d_in[2];
  // d_in[3] = conv bias: mathematically cancels under training-mode BatchNorm
  const float* gamma = (const float*)d_in[4];
  const float* beta  = (const float*)d_in[5];
  float* out = (float*)d_out;

  const int E = in_sizes[0] / 64;   // 1,000,000 (divisible by 64)

  __bf16* Wb   = (__bf16*)d_ws;
  float* stats = (float*)((char*)d_ws + 81920);
  float* gsum  = stats;
  float* gsq   = stats + 128;
  float* scale = stats + 256;
  float* shift = stats + 384;

  prep_kernel <<<(OUT_CH * K_TOT + 255) / 256, 256, 0, stream>>>(W, Wb, stats);
  conv_kernel <<<E / M_TILE, 256, 0, stream>>>(x, e2e, Wb, out, gsum, gsq);
  stats_kernel<<<1, 128, 0, stream>>>(gsum, gsq, gamma, beta, scale, shift,
                                      1.0f / (float)E);
  bnrelu_kernel<<<(int)(((long)E * OUT_CH / 4) / 256), 256, 0, stream>>>(
      (float4*)out, scale, shift);
}
